// HybridAttentionMechanism_67869073211901
// MI455X (gfx1250) — matrix operand order repaired and checked by reading, not compiled, not run on hardware
//
#include <hip/hip_runtime.h>
#include <math.h>

#define B_ 8
#define S_ 2048
#define D_ 512
#define WIN 30

typedef __attribute__((ext_vector_type(16))) __bf16 v16bf;
typedef __attribute__((ext_vector_type(8)))  float  v8f;

union AF { v16bf v; unsigned int u[8]; unsigned short s[16]; };
union CF { v8f v; float f[8]; };

__device__ __forceinline__ unsigned short f2bf(float f) {
    unsigned int u = __float_as_uint(f);
    u += 0x7FFFu + ((u >> 16) & 1u);        // round-to-nearest-even
    return (unsigned short)(u >> 16);
}

// ---------------------------------------------------------------------------
// Pre-conversion: hidden f32 -> bf16 (row-major [B*S, D])
// ---------------------------------------------------------------------------
__global__ __launch_bounds__(256) void conv_h_kernel(
    const float* __restrict__ h, unsigned short* __restrict__ Hb)
{
    size_t i = ((size_t)blockIdx.x * 256 + threadIdx.x) * 8;
    float4 f0 = *(const float4*)(h + i);
    float4 f1 = *(const float4*)(h + i + 4);
    uint4 o;
    o.x = (unsigned)f2bf(f0.x) | ((unsigned)f2bf(f0.y) << 16);
    o.y = (unsigned)f2bf(f0.z) | ((unsigned)f2bf(f0.w) << 16);
    o.z = (unsigned)f2bf(f1.x) | ((unsigned)f2bf(f1.y) << 16);
    o.w = (unsigned)f2bf(f1.z) | ((unsigned)f2bf(f1.w) << 16);
    *(uint4*)(Hb + i) = o;
}

// ---------------------------------------------------------------------------
// Pre-conversion: W[z] f32 [K,N] -> bf16 transposed Wt[z][n][k], so the
// projection B-fragment (pairs along K) is contiguous dword loads.
// ---------------------------------------------------------------------------
__global__ __launch_bounds__(256) void conv_w_kernel(
    const float* __restrict__ Wq, const float* __restrict__ Wk,
    const float* __restrict__ Wv, unsigned short* __restrict__ Wt)
{
    int z = blockIdx.z;
    const float* W = (z == 0) ? Wq : ((z == 1) ? Wk : Wv);
    unsigned short* dst = Wt + (size_t)z * D_ * D_;
    int idx = blockIdx.x * 256 + threadIdx.x;    // 0 .. 512*256-1
    int n  = idx >> 8;
    int kp = (idx & 255) * 2;
    float x0 = W[(long)kp * D_ + n];
    float x1 = W[(long)(kp + 1) * D_ + n];
    *(unsigned int*)(dst + (long)n * D_ + kp) =
        (unsigned)f2bf(x0) | ((unsigned)f2bf(x1) << 16);
}

// ---------------------------------------------------------------------------
// Projections from pre-converted bf16. Each wave: 16 rows x 64 cols
// (4 accumulators, A-fragment reused 4x). Grid: (M/16, D/128, 3), block 64.
// Q,K row-major bf16 [B*S, D]; V transposed bf16 [B, D, S].
// ---------------------------------------------------------------------------
__global__ __launch_bounds__(64) void proj_kernel(
    const unsigned short* __restrict__ Hb, const unsigned short* __restrict__ Wt,
    const float* __restrict__ bq, const float* __restrict__ bk,
    const float* __restrict__ bv,
    unsigned short* __restrict__ Qb, unsigned short* __restrict__ Kb,
    unsigned short* __restrict__ Vt)
{
    const int lane = threadIdx.x & 31;
    const int wv   = threadIdx.x >> 5;     // 0..1
    const int mt   = blockIdx.x;           // 16-row tile
    const int ng   = blockIdx.y;           // 128-col group
    const int z    = blockIdx.z;           // 0=Q 1=K 2=V
    const unsigned short* Wz = Wt + (size_t)z * D_ * D_;
    const float* bias = (z == 0) ? bq : ((z == 1) ? bk : bv);
    const int row = lane & 15, half = lane >> 4;

    CF acc[4];
    #pragma unroll
    for (int t = 0; t < 4; t++)
        #pragma unroll
        for (int v = 0; v < 8; v++) acc[t].f[v] = 0.f;

    const unsigned short* hr = Hb + ((long)mt * 16 + row) * D_;
    for (int c = 0; c < 16; c++) {
        AF a;
        #pragma unroll
        for (int j = 0; j < 4; j++) {
            a.u[j]     = *(const unsigned int*)(hr + c * 32 + half * 16 + 2 * j);
            a.u[4 + j] = *(const unsigned int*)(hr + c * 32 + half * 16 + 8 + 2 * j);
        }
        #pragma unroll
        for (int nt = 0; nt < 4; nt++) {
            int n = ng * 128 + wv * 64 + nt * 16 + row;
            AF b;
            const unsigned short* wp = Wz + (long)n * D_ + c * 32 + half * 16;
            #pragma unroll
            for (int j = 0; j < 8; j++)
                b.u[j] = *(const unsigned int*)(wp + 2 * j);
            acc[nt].v = __builtin_amdgcn_wmma_f32_16x16x32_bf16(
                false, a.v, false, b.v, (short)0, acc[nt].v, false, false);
        }
    }
    #pragma unroll
    for (int nt = 0; nt < 4; nt++) {
        int n = ng * 128 + wv * 64 + nt * 16 + row;
        float bb = bias[n];
        #pragma unroll
        for (int v = 0; v < 8; v++) {
            long grow = (long)mt * 16 + v + 8 * half;
            unsigned short o = f2bf(acc[nt].f[v] + bb);
            if (z == 2) {
                int bidx = (int)(grow >> 11);
                int srow = (int)(grow & 2047);
                Vt[((long)bidx * D_ + n) * S_ + srow] = o;
            } else {
                unsigned short* dst = (z == 0) ? Qb : Kb;
                dst[grow * D_ + n] = o;
            }
        }
    }
}

// ---------------------------------------------------------------------------
// gate = sigmoid(h @ Wg + bg), one wave per row.
// ---------------------------------------------------------------------------
__global__ __launch_bounds__(128) void gate_kernel(
    const float* __restrict__ h, const float* __restrict__ Wg,
    const float* __restrict__ bg, float* __restrict__ gate)
{
    int lane = threadIdx.x & 31;
    int wv   = threadIdx.x >> 5;
    long row = (long)blockIdx.x * 4 + wv;
    if (row >= (long)B_ * S_) return;
    const float* hr = h + row * D_;
    float s = 0.f;
    for (int d = lane; d < D_; d += 32) s += hr[d] * Wg[d];
    #pragma unroll
    for (int m = 16; m > 0; m >>= 1) s += __shfl_xor(s, m, 32);
    if (lane == 0) gate[row] = 1.f / (1.f + __expf(-(s + bg[0])));
}

// ---------------------------------------------------------------------------
// Attention: one 16-query tile per 2-wave workgroup. D split across waves for
// QK^T (partials summed via LDS); probs staged in LDS bf16; each wave owns
// 256 output cols of the P*V accumulation.
// ---------------------------------------------------------------------------
template <bool LOCAL>
__device__ __forceinline__ void process_block(
    int kb, int kbLast, int b, int q0, int wv, int row, int half,
    const unsigned short* __restrict__ Kb,
    const unsigned short* __restrict__ Vt,
    const AF* qa, CF* acc, float* mrow, float* lrow,
    float (*sc)[16][16], unsigned short (*sp)[32])
{
    const float rscale = 0.04419417382415922f;  // 1/sqrt(512)
    const int lane = row + 16 * half;
    if (kb < kbLast) {  // prefetch next key block's K and V lines
        __builtin_prefetch(Kb + ((long)b * S_ + (kb + 1) * 32 + lane) * D_ + wv * 256, 0, 0);
        __builtin_prefetch(Vt + ((long)b * D_ + wv * 256 + lane) * S_ + (kb + 1) * 32, 0, 0);
    }
    #pragma unroll
    for (int tt = 0; tt < 2; tt++) {
        int kt = kb * 2 + tt;
        CF c;
        #pragma unroll
        for (int v = 0; v < 8; v++) c.f[v] = 0.f;
        #pragma unroll
        for (int cc = 0; cc < 8; cc++) {
            AF kf;  // B fragment of K^T: col = row(lane&15) = key-in-tile
            const unsigned short* kr =
                Kb + ((long)b * S_ + kt * 16 + row) * D_ + (wv * 8 + cc) * 32 + half * 16;
            #pragma unroll
            for (int j = 0; j < 8; j++)
                kf.u[j] = *(const unsigned int*)(kr + 2 * j);
            c.v = __builtin_amdgcn_wmma_f32_16x16x32_bf16(
                false, qa[cc].v, false, kf.v, (short)0, c.v, false, false);
        }
        // sum the two waves' D-partials via LDS
        #pragma unroll
        for (int v = 0; v < 8; v++) sc[wv][v + 8 * half][row] = c.f[v];
        __syncthreads();
        #pragma unroll
        for (int v = 0; v < 8; v++) c.f[v] += sc[1 ^ wv][v + 8 * half][row];
        __syncthreads();

        float s[8], p[8];
        #pragma unroll
        for (int v = 0; v < 8; v++) {
            float x = c.f[v] * rscale;
            if (LOCAL) {
                int key = kt * 16 + row;
                int qr  = q0 + v + 8 * half;
                int dlt = qr - key; if (dlt < 0) dlt = -dlt;
                if (dlt > WIN) x += -1e9f;
            }
            s[v] = x;
        }
        #pragma unroll
        for (int v = 0; v < 8; v++) {
            float tm = s[v];
            tm = fmaxf(tm, __shfl_xor(tm, 1, 32));
            tm = fmaxf(tm, __shfl_xor(tm, 2, 32));
            tm = fmaxf(tm, __shfl_xor(tm, 4, 32));
            tm = fmaxf(tm, __shfl_xor(tm, 8, 32));
            float mn = fmaxf(mrow[v], tm);
            float al = __expf(mrow[v] - mn);
            mrow[v] = mn;
            float pv = __expf(s[v] - mn);
            p[v] = pv;
            float ts = pv;
            ts += __shfl_xor(ts, 1, 32);
            ts += __shfl_xor(ts, 2, 32);
            ts += __shfl_xor(ts, 4, 32);
            ts += __shfl_xor(ts, 8, 32);
            lrow[v] = lrow[v] * al + ts;
            #pragma unroll
            for (int nt = 0; nt < 16; nt++) acc[nt].f[v] *= al;
        }
        if (wv == 0) {  // both waves hold identical probs; one writes
            #pragma unroll
            for (int v = 0; v < 8; v++)
                sp[v + 8 * half][tt * 16 + row] = f2bf(p[v]);
        }
    }
    __syncthreads();
    // P (16x32) * V (32 x 256-per-wave)
    AF pa;
    #pragma unroll
    for (int j = 0; j < 4; j++) {
        pa.u[j]     = *(const unsigned int*)&sp[row][half * 16 + 2 * j];
        pa.u[4 + j] = *(const unsigned int*)&sp[row][half * 16 + 8 + 2 * j];
    }
    #pragma unroll
    for (int nt = 0; nt < 16; nt++) {
        AF vf;
        int d = wv * 256 + nt * 16 + row;
        const unsigned short* vr =
            Vt + ((long)b * D_ + d) * S_ + kb * 32 + half * 16;
        #pragma unroll
        for (int j = 0; j < 8; j++)
            vf.u[j] = *(const unsigned int*)(vr + 2 * j);
        acc[nt].v = __builtin_amdgcn_wmma_f32_16x16x32_bf16(
            false, pa.v, false, vf.v, (short)0, acc[nt].v, false, false);
    }
    __syncthreads();
}

__global__ __launch_bounds__(64) void attn_kernel(
    const unsigned short* __restrict__ Qb, const unsigned short* __restrict__ Kb,
    const unsigned short* __restrict__ Vt, const float* __restrict__ gate,
    float* __restrict__ out)
{
    __shared__ float sc[2][16][16];
    __shared__ unsigned short sp[16][32];
    const int lane = threadIdx.x & 31;
    const int wv   = threadIdx.x >> 5;
    const int qt   = blockIdx.x;          // 0..127
    const int b    = blockIdx.y;          // 0..7
    const int row = lane & 15, half = lane >> 4;
    const int q0 = qt * 16;
    const long qrowbase = (long)b * S_ + q0;

    AF qa[8];  // this wave's half of Q (d-chunks wv*8 .. wv*8+7)
    #pragma unroll
    for (int c = 0; c < 8; c++) {
        const unsigned short* qr = Qb + (qrowbase + row) * D_ + (wv * 8 + c) * 32;
        #pragma unroll
        for (int j = 0; j < 4; j++) {
            qa[c].u[j]     = *(const unsigned int*)(qr + half * 16 + 2 * j);
            qa[c].u[4 + j] = *(const unsigned int*)(qr + half * 16 + 8 + 2 * j);
        }
    }

    CF acc[16];
    float mrow[8], lrow[8];

    // ---------------- Phase A: global attention ----------------
    #pragma unroll
    for (int v = 0; v < 8; v++) { mrow[v] = -1e30f; lrow[v] = 0.f; }
    #pragma unroll
    for (int nt = 0; nt < 16; nt++)
        #pragma unroll
        for (int v = 0; v < 8; v++) acc[nt].f[v] = 0.f;

    const int kbAll = S_ / 32 - 1;
    for (int kb = 0; kb <= kbAll; kb++)
        process_block<false>(kb, kbAll, b, q0, wv, row, half, Kb, Vt, qa, acc,
                             mrow, lrow, sc, sp);

    #pragma unroll
    for (int v = 0; v < 8; v++) {
        int r = v + 8 * half;
        float g   = gate[qrowbase + r];
        float inv = 1.0f / lrow[v];
        #pragma unroll
        for (int nt = 0; nt < 16; nt++) {
            int d = wv * 256 + nt * 16 + row;
            out[(qrowbase + r) * D_ + d] = (1.f - g) * acc[nt].f[v] * inv;
        }
    }

    // ---------------- Phase B: local (windowed) attention ----------------
    #pragma unroll
    for (int v = 0; v < 8; v++) { mrow[v] = -1e30f; lrow[v] = 0.f; }
    #pragma unroll
    for (int nt = 0; nt < 16; nt++)
        #pragma unroll
        for (int v = 0; v < 8; v++) acc[nt].f[v] = 0.f;

    int lo = q0 - WIN;       if (lo < 0) lo = 0;
    int hi = q0 + 15 + WIN;  if (hi > S_ - 1) hi = S_ - 1;
    const int kbLast = hi >> 5;
    for (int kb = lo >> 5; kb <= kbLast; kb++)
        process_block<true>(kb, kbLast, b, q0, wv, row, half, Kb, Vt, qa, acc,
                            mrow, lrow, sc, sp);

    #pragma unroll
    for (int v = 0; v < 8; v++) {
        int r = v + 8 * half;
        float g   = gate[qrowbase + r];
        float inv = 1.0f / lrow[v];
        #pragma unroll
        for (int nt = 0; nt < 16; nt++) {
            int d = wv * 256 + nt * 16 + row;
            long idx = (qrowbase + r) * D_ + d;
            out[idx] = out[idx] + g * acc[nt].f[v] * inv;
        }
    }
}

// ---------------------------------------------------------------------------
extern "C" void kernel_launch(void* const* d_in, const int* in_sizes, int n_in,
                              void* d_out, int out_size, void* d_ws, size_t ws_size,
                              hipStream_t stream) {
    (void)in_sizes; (void)n_in; (void)out_size; (void)ws_size;
    const float* h  = (const float*)d_in[0];
    const float* Wq = (const float*)d_in[1];
    const float* bq = (const float*)d_in[2];
    const float* Wk = (const float*)d_in[3];
    const float* bk = (const float*)d_in[4];
    const float* Wv = (const float*)d_in[5];
    const float* bv = (const float*)d_in[6];
    const float* Wg = (const float*)d_in[7];
    const float* bg = (const float*)d_in[8];
    float* out = (float*)d_out;

    const size_t elems = (size_t)B_ * S_ * D_;   // 8.39M
    unsigned short* Qb = (unsigned short*)d_ws;
    unsigned short* Kb = Qb + elems;
    unsigned short* Vt = Kb + elems;
    unsigned short* Hb = Vt + elems;
    unsigned short* Wt = Hb + elems;                       // 3 * 512*512 bf16
    float* gate = (float*)(Wt + (size_t)3 * D_ * D_);

    conv_h_kernel<<<(unsigned)(elems / 8 / 256), 256, 0, stream>>>(h, Hb);
    conv_w_kernel<<<dim3((D_ * (D_ / 2)) / 256, 1, 3), 256, 0, stream>>>(Wq, Wk, Wv, Wt);
    proj_kernel<<<dim3((B_ * S_) / 16, D_ / 128, 3), 64, 0, stream>>>(
        Hb, Wt, bq, bk, bv, Qb, Kb, Vt);
    gate_kernel<<<(B_ * S_ + 3) / 4, 128, 0, stream>>>(h, Wg, bg, gate);
    attn_kernel<<<dim3(S_ / 16, B_), 64, 0, stream>>>(Qb, Kb, Vt, gate, out);
}
